// MoeLayer_38955353374969
// MI455X (gfx1250) — matrix operand order repaired and checked by reading, not compiled, not run on hardware
//
#include <hip/hip_runtime.h>
#include <hip/hip_bf16.h>

// MoE top-2 of 8, T=4096 tokens, D=2048, H=4096.
// Strategy: routed grouped-GEMM, bf16 WMMA (v_wmma_f32_16x16x32_bf16) with f32
// accumulate; fp32 weights converted to bf16 in-register while staging to LDS.
// Pass-B A-tiles (bf16 h rows) are moved with GLOBAL_LOAD_ASYNC_TO_LDS_B128
// (ASYNCcnt) when the toolchain exposes the builtin.
// Workspace layout (needs ~76 MB):
//   [0..32)        counts[8]        (int)
//   [64..100)      bases[9]         (int, padded exclusive scan)
//   [4096..)       tok_idx[8*4096]  (int)
//   [+128KB..)     tok_w [8*4096]   (float)
//   [512KB..)      h buffer, bf16, <= 9216 rows x 4096

#define T_TOK 4096
#define DDIM  2048
#define HDIM  4096
#define NEXP  8

#define MT 128   // rows (token-expert pairs) per block
#define NT 128   // output cols per block
#define KT 32    // K per WMMA step
#define LDT 40   // padded LDS row stride in halves (80B, 16B-aligned)

typedef __bf16 bf16;
typedef __attribute__((ext_vector_type(16))) __bf16 v16bf;
typedef __attribute__((ext_vector_type(8)))  __bf16 v8bf;
typedef __attribute__((ext_vector_type(8)))  float  v8f;
typedef __attribute__((ext_vector_type(4)))  int    v4i;

#define AS1 __attribute__((address_space(1)))
#define AS3 __attribute__((address_space(3)))

typedef AS1 v4i* g_v4i_p;   // global int4* (matches builtin param 0)
typedef AS3 v4i* l_v4i_p;   // LDS int4*    (matches builtin param 1)

#if defined(__has_builtin)
#  if __has_builtin(__builtin_amdgcn_global_load_async_to_lds_b128)
#    define HAVE_ASYNC_COPY 1
#  endif
#  if __has_builtin(__builtin_amdgcn_s_wait_asynccnt)
#    define HAVE_WAIT_ASYNC 1
#  endif
#endif

__device__ inline void wait_asynccnt0() {
#if defined(HAVE_WAIT_ASYNC)
  __builtin_amdgcn_s_wait_asynccnt(0);
#else
  asm volatile("s_wait_asynccnt 0x0" ::: "memory");
#endif
}

// ---------------- utility kernels ----------------

__global__ void moe_zero(float* __restrict__ out, size_t n, int* __restrict__ counts) {
  size_t i = (size_t)blockIdx.x * blockDim.x + threadIdx.x;
  size_t stride = (size_t)gridDim.x * blockDim.x;
  for (; i < n; i += stride) out[i] = 0.0f;
  if (blockIdx.x == 0 && threadIdx.x < NEXP) counts[threadIdx.x] = 0;
}

// One 128-thread block per token: 8 dot products of length D, top-2 + softmax,
// append (token, weight) to the selected experts' lists.
__global__ __launch_bounds__(128)
void moe_gate(const float* __restrict__ x, const float* __restrict__ gw,
              int* __restrict__ counts, int* __restrict__ tok_idx,
              float* __restrict__ tok_w) {
  const int t = blockIdx.x;
  const float* xr = x + (size_t)t * DDIM;
  float acc[NEXP];
#pragma unroll
  for (int e = 0; e < NEXP; ++e) acc[e] = 0.0f;
  for (int i = threadIdx.x; i < DDIM; i += blockDim.x) {
    float xv = xr[i];
#pragma unroll
    for (int e = 0; e < NEXP; ++e) acc[e] = fmaf(xv, gw[e * DDIM + i], acc[e]);
  }
  __shared__ float sred[NEXP];
  if (threadIdx.x < NEXP) sred[threadIdx.x] = 0.0f;
  __syncthreads();
#pragma unroll
  for (int e = 0; e < NEXP; ++e) atomicAdd(&sred[e], acc[e]);
  __syncthreads();
  if (threadIdx.x == 0) {
    float l[NEXP];
#pragma unroll
    for (int e = 0; e < NEXP; ++e) l[e] = sred[e];
    int e0 = 0;
#pragma unroll
    for (int e = 1; e < NEXP; ++e) if (l[e] > l[e0]) e0 = e;
    int e1 = (e0 == 0) ? 1 : 0;
#pragma unroll
    for (int e = 0; e < NEXP; ++e) { if (e != e0 && l[e] > l[e1]) e1 = e; }
    float m  = fmaxf(l[e0], l[e1]);
    float w0 = __expf(l[e0] - m), w1 = __expf(l[e1] - m);
    float inv = 1.0f / (w0 + w1);
    w0 *= inv; w1 *= inv;
    int p0 = atomicAdd(&counts[e0], 1);
    tok_idx[e0 * T_TOK + p0] = t; tok_w[e0 * T_TOK + p0] = w0;
    int p1 = atomicAdd(&counts[e1], 1);
    tok_idx[e1 * T_TOK + p1] = t; tok_w[e1 * T_TOK + p1] = w1;
  }
}

// Padded exclusive scan of per-expert counts -> compacted h-row bases.
__global__ void moe_scan(const int* __restrict__ counts, int* __restrict__ bases) {
  if (threadIdx.x == 0) {
    int b = 0;
    for (int e = 0; e < NEXP; ++e) {
      bases[e] = b;
      b += ((counts[e] + MT - 1) / MT) * MT;
    }
    bases[NEXP] = b;
  }
}

// ---------------- WMMA fragment loaders (ISA 7.12.2 layouts, wave32) --------

// A (16x32 bf16): lane l: M = l&15; lanes 0-15 hold K 0-7 / 16-23 in vector
// elems 0-7 / 8-15; lanes 16-31 hold K 8-15 / 24-31.
__device__ inline v16bf lds_a_frag(const bf16* base, int m0) {
  const int l = threadIdx.x & 31;
  const int l16 = l & 15, hi = l >> 4;
  const bf16* p = base + (m0 + l16) * LDT + hi * 8;
  v8bf a0 = *(const v8bf*)(p);        // K = hi*8 + 0..7
  v8bf a1 = *(const v8bf*)(p + 16);   // K = 16 + hi*8 + 0..7
  v16bf r;
#pragma unroll
  for (int j = 0; j < 8; ++j) { r[j] = a0[j]; r[j + 8] = a1[j]; }
  return r;
}

// B (32x16 bf16) from a transposed [n][k] LDS tile: lane l: N = l&15;
// lanes 0-15 hold K 0-7 / 16-23, lanes 16-31 hold K 8-15 / 24-31 (same as A).
__device__ inline v16bf lds_b_frag(const bf16* base, int n0) {
  const int l = threadIdx.x & 31;
  const int l16 = l & 15, hi = l >> 4;
  const bf16* p = base + (n0 + l16) * LDT + hi * 8;
  v8bf a0 = *(const v8bf*)(p);
  v8bf a1 = *(const v8bf*)(p + 16);
  v16bf r;
#pragma unroll
  for (int j = 0; j < 8; ++j) { r[j] = a0[j]; r[j + 8] = a1[j]; }
  return r;
}

// ---------------- pass A: h = silu(x@w1) * (x@w3), gathered rows ------------

__global__ __launch_bounds__(256)
void moe_up_gemm(const float* __restrict__ x, const float* __restrict__ w1,
                 const float* __restrict__ w3, const int* __restrict__ counts,
                 const int* __restrict__ bases, const int* __restrict__ tok_idx,
                 bf16* __restrict__ h) {
  const int e = blockIdx.z;
  const int cnt = counts[e];
  const int padded = ((cnt + MT - 1) / MT) * MT;
  const int m0 = blockIdx.x * MT;
  if (m0 >= padded) return;               // block-uniform: EXEC stays all-1
  const int n0 = blockIdx.y * NT;

  __shared__ bf16 sA [MT][LDT];
  __shared__ bf16 sB1[NT][LDT];
  __shared__ bf16 sB3[NT][LDT];
  __shared__ int  sTok[MT];

  const int tid = threadIdx.x;
  if (tid < MT) {
    int m = m0 + tid;
    sTok[tid] = (m < cnt) ? tok_idx[e * T_TOK + m] : 0;  // pad rows reuse token 0
  }
  __syncthreads();

  // Hoisted gather pointer for this thread's A-staging row (loop-invariant).
  const int arow = tid >> 1;
  const int ac0  = (tid & 1) * 16;
  const float* asrc = x + (size_t)sTok[arow] * DDIM + ac0;
  bf16* adst = &sA[arow][ac0];

  const int wave = tid >> 5;
  const int wm = wave >> 2;   // 0..1 -> M offset wm*64
  const int wn = wave & 3;    // 0..3 -> N offset wn*32

  v8f acc1[4][2], acc3[4][2];
#pragma unroll
  for (int i = 0; i < 4; ++i)
#pragma unroll
    for (int j = 0; j < 2; ++j)
#pragma unroll
      for (int v = 0; v < 8; ++v) { acc1[i][j][v] = 0.0f; acc3[i][j][v] = 0.0f; }

  const float* w1e = w1 + (size_t)e * DDIM * HDIM;
  const float* w3e = w3 + (size_t)e * DDIM * HDIM;

  for (int k0 = 0; k0 < DDIM; k0 += KT) {
    __syncthreads();
    {  // stage A: 128 rows x 32 halves, gathered token rows, fp32 -> bf16
      const float* src = asrc + k0;
#pragma unroll
      for (int j = 0; j < 16; ++j) adst[j] = (bf16)src[j];
    }
    {  // stage B1/B3 transposed: [n][k], fp32 -> bf16
      const int kk = tid >> 3;
      const int c0 = (tid & 7) * 16;
      const float* s1 = w1e + (size_t)(k0 + kk) * HDIM + n0 + c0;
      const float* s3 = w3e + (size_t)(k0 + kk) * HDIM + n0 + c0;
#pragma unroll
      for (int j = 0; j < 16; ++j) {
        sB1[c0 + j][kk] = (bf16)s1[j];
        sB3[c0 + j][kk] = (bf16)s3[j];
      }
    }
    __syncthreads();

    if (k0 + KT < DDIM) {  // pull next weight tile toward L2 (global_prefetch_b8)
      const float* pf = w1e + (size_t)(k0 + KT + (tid >> 3)) * HDIM + n0 + (tid & 7) * 16;
      __builtin_prefetch(pf, 0, 1);
    }

    v16bf afr[4];
#pragma unroll
    for (int mi = 0; mi < 4; ++mi) afr[mi] = lds_a_frag(&sA[0][0], wm * 64 + mi * 16);
#pragma unroll
    for (int ni = 0; ni < 2; ++ni) {
      v16bf b1 = lds_b_frag(&sB1[0][0], wn * 32 + ni * 16);
      v16bf b3 = lds_b_frag(&sB3[0][0], wn * 32 + ni * 16);
#pragma unroll
      for (int mi = 0; mi < 4; ++mi) {
        acc1[mi][ni] = __builtin_amdgcn_wmma_f32_16x16x32_bf16(
            false, afr[mi], false, b1, (short)0, acc1[mi][ni], false, false);
        acc3[mi][ni] = __builtin_amdgcn_wmma_f32_16x16x32_bf16(
            false, afr[mi], false, b3, (short)0, acc3[mi][ni], false, false);
      }
    }
  }

  // Epilogue: SwiGLU, store bf16 h rows. C/D layout: VGPR v, lane l -> M = v + (l>=16)*8, N = l&15.
  const int l = tid & 31, l16 = l & 15, hi = l >> 4;
  const int row_base = bases[e] + m0;
#pragma unroll
  for (int mi = 0; mi < 4; ++mi) {
#pragma unroll
    for (int ni = 0; ni < 2; ++ni) {
#pragma unroll
      for (int v = 0; v < 8; ++v) {
        const int m = wm * 64 + mi * 16 + hi * 8 + v;
        const int n = n0 + wn * 32 + ni * 16 + l16;
        const float a = acc1[mi][ni][v];
        const float g = a / (1.0f + __expf(-a));      // silu
        h[(size_t)(row_base + m) * HDIM + n] = (bf16)(g * acc3[mi][ni][v]);
      }
    }
  }
}

// ---------------- pass B: out[t] += w * (h @ w2[e]) -------------------------

__global__ __launch_bounds__(256)
void moe_down_gemm(const float* __restrict__ w2, const int* __restrict__ counts,
                   const int* __restrict__ bases, const int* __restrict__ tok_idx,
                   const float* __restrict__ tok_w, const bf16* __restrict__ h,
                   float* __restrict__ out) {
  const int e = blockIdx.z;
  const int cnt = counts[e];
  const int padded = ((cnt + MT - 1) / MT) * MT;
  const int m0 = blockIdx.x * MT;
  if (m0 >= padded) return;
  const int n0 = blockIdx.y * NT;   // over DDIM

  __shared__ bf16 sA[MT][LDT];
  __shared__ bf16 sB[NT][LDT];
  __shared__ int   sTok[MT];
  __shared__ float sW[MT];

  const int tid = threadIdx.x;
  if (tid < MT) {
    int m = m0 + tid;
    bool valid = m < cnt;
    sTok[tid] = valid ? tok_idx[e * T_TOK + m] : 0;
    sW[tid]   = valid ? tok_w[e * T_TOK + m] : 0.0f;
  }

  const int wave = tid >> 5;
  const int wm = wave >> 2, wn = wave & 3;

  v8f acc[4][2];
#pragma unroll
  for (int i = 0; i < 4; ++i)
#pragma unroll
    for (int j = 0; j < 2; ++j)
#pragma unroll
      for (int v = 0; v < 8; ++v) acc[i][j][v] = 0.0f;

  const float* w2e = w2 + (size_t)e * HDIM * DDIM;
  const bf16* hrow = h + (size_t)(bases[e] + m0) * HDIM;

  // This thread's A-staging addresses (h rows are already bf16 -> raw DMA copy).
  const int arow = tid >> 1;
  const int ac0  = (tid & 1) * 16;
  const bf16* asrc = hrow + (size_t)arow * HDIM + ac0;
  bf16* adst = &sA[arow][ac0];

  for (int k0 = 0; k0 < HDIM; k0 += KT) {
    __syncthreads();
    {  // stage A from h: async global->LDS DMA (ASYNCcnt) when available
      const bf16* src = asrc + k0;
#if defined(HAVE_ASYNC_COPY)
      __builtin_amdgcn_global_load_async_to_lds_b128(
          (g_v4i_p)(uintptr_t)src, (l_v4i_p)(uintptr_t)adst, 0, 0);
      __builtin_amdgcn_global_load_async_to_lds_b128(
          (g_v4i_p)(uintptr_t)(src + 8), (l_v4i_p)(uintptr_t)(adst + 8), 0, 0);
#else
      const v8bf* s = (const v8bf*)src;
      v8bf* d = (v8bf*)adst;
      d[0] = s[0];
      d[1] = s[1];
#endif
    }
    {  // stage w2 transposed, fp32 -> bf16
      const int kk = tid >> 3;
      const int c0 = (tid & 7) * 16;
      const float* s2 = w2e + (size_t)(k0 + kk) * DDIM + n0 + c0;
#pragma unroll
      for (int j = 0; j < 16; ++j) sB[c0 + j][kk] = (bf16)s2[j];
    }
#if defined(HAVE_ASYNC_COPY)
    wait_asynccnt0();   // A-tile DMA complete before the barrier
#endif
    __syncthreads();

    if (k0 + KT < HDIM) {
      const float* pf = w2e + (size_t)(k0 + KT + (tid >> 3)) * DDIM + n0 + (tid & 7) * 16;
      __builtin_prefetch(pf, 0, 1);
    }

    v16bf afr[4];
#pragma unroll
    for (int mi = 0; mi < 4; ++mi) afr[mi] = lds_a_frag(&sA[0][0], wm * 64 + mi * 16);
#pragma unroll
    for (int ni = 0; ni < 2; ++ni) {
      v16bf b = lds_b_frag(&sB[0][0], wn * 32 + ni * 16);
#pragma unroll
      for (int mi = 0; mi < 4; ++mi) {
        acc[mi][ni] = __builtin_amdgcn_wmma_f32_16x16x32_bf16(
            false, afr[mi], false, b, (short)0, acc[mi][ni], false, false);
      }
    }
  }

  // Epilogue: weighted scatter-add into out (each token appears in 2 experts).
  const int l = tid & 31, l16 = l & 15, hi = l >> 4;
#pragma unroll
  for (int mi = 0; mi < 4; ++mi) {
#pragma unroll
    for (int ni = 0; ni < 2; ++ni) {
#pragma unroll
      for (int v = 0; v < 8; ++v) {
        const int m = wm * 64 + mi * 16 + hi * 8 + v;
        const float w = sW[m];
        if (w != 0.0f) {
          const int t = sTok[m];
          const int n = n0 + wn * 32 + ni * 16 + l16;
          atomicAdd(&out[(size_t)t * DDIM + n], w * acc[mi][ni][v]);
        }
      }
    }
  }
}

// ---------------- launch ----------------

extern "C" void kernel_launch(void* const* d_in, const int* in_sizes, int n_in,
                              void* d_out, int out_size, void* d_ws, size_t ws_size,
                              hipStream_t stream) {
  const float* x  = (const float*)d_in[0];   // [B,S,D]
  const float* gw = (const float*)d_in[1];   // [E,D]
  const float* w1 = (const float*)d_in[2];   // [E,D,H]
  const float* w3 = (const float*)d_in[3];   // [E,D,H]
  const float* w2 = (const float*)d_in[4];   // [E,H,D]
  float* out = (float*)d_out;

  char* ws = (char*)d_ws;
  int*   counts  = (int*)(ws);
  int*   bases   = (int*)(ws + 64);
  int*   tok_idx = (int*)(ws + 4096);
  float* tok_w   = (float*)(ws + 4096 + (size_t)NEXP * T_TOK * 4);
  bf16*  h       = (bf16*)(ws + (1 << 19));   // ~75.5 MB max (9216 x 4096 bf16)

  moe_zero<<<512, 256, 0, stream>>>(out, (size_t)T_TOK * DDIM, counts);
  moe_gate<<<T_TOK, 128, 0, stream>>>(x, gw, counts, tok_idx, tok_w);
  moe_scan<<<1, 32, 0, stream>>>(counts, bases);
  moe_up_gemm<<<dim3(T_TOK / MT, HDIM / NT, NEXP), 256, 0, stream>>>(
      x, w1, w3, counts, bases, tok_idx, h);
  moe_down_gemm<<<dim3(T_TOK / MT, DDIM / NT, NEXP), 256, 0, stream>>>(
      w2, counts, bases, tok_idx, tok_w, h, out);
}
